// DeltaNet_31877247271428
// MI455X (gfx1250) — matrix operand order repaired and checked by reading, not compiled, not run on hardware
//
#include <hip/hip_runtime.h>
#include <hip/hip_bf16.h>
#include <math.h>

// ---------------------------------------------------------------------------
// DeltaNet layer for MI455X (gfx1250, wave32, WMMA).
// All dense GEMMs use v_wmma_f32_16x16x32_f16 (f16 mul, f32 acc); N is a
// template parameter (folds all B strides to immediates), K/act/bias are
// runtime (K only affects one-time pointer setup; staging pointers advance by
// constant strides).  All LDS tiles are stored PRE-SWIZZLED in WMMA fragment
// order so fragment loads are ds_load_b128; f32->f16 via v_cvt_pk_rtz_f16_f32.
// The sequential delta-rule scan keeps the full 256x256 f32 state S resident
// in LDS (256KB of the 320KB/WGP budget -- impossible on CDNA4) in
// B-fragment-major order for all 128 chunks.
// ---------------------------------------------------------------------------

typedef __attribute__((ext_vector_type(16))) _Float16 v16h;
typedef __attribute__((ext_vector_type(8)))  _Float16 v8h;
typedef __attribute__((ext_vector_type(2)))  __fp16   v2hf;   // cvt_pkrtz result type
typedef __attribute__((ext_vector_type(8)))  float    v8f;
typedef __attribute__((ext_vector_type(4)))  float    f4v;

#define B_   2
#define L_   4096
#define D_   1024
#define H_   4
#define DV_  256
#define CHK_ 32
#define NC_  128           // L_/CHK_
#define BL_  (B_*L_)       // 8192
#define BLD_ ((size_t)BL_*D_)

// ------------------------------ math helpers -------------------------------
static __device__ __forceinline__ float siluf(float x)   { return x / (1.f + __expf(-x)); }
static __device__ __forceinline__ float sigmf(float x)   { return 1.f / (1.f + __expf(-x)); }
static __device__ __forceinline__ float geluf(float x)   { return 0.5f * x * (1.f + erff(x * 0.70710678118654752f)); }

static __device__ __forceinline__ v16h cat8(v8h lo, v8h hi) {
  return __builtin_shufflevector(lo, hi, 0,1,2,3,4,5,6,7,8,9,10,11,12,13,14,15);
}
static __device__ __forceinline__ v8f wmma16(v16h a, v16h b, v8f c) {
  return __builtin_amdgcn_wmma_f32_16x16x32_f16(false, a, false, b, (short)0, c, false, false);
}
// packed f32x8 -> f16x8 conversion (v_cvt_pk_rtz_f16_f32)
static __device__ __forceinline__ v8h pk8(f4v a0, f4v a1) {
#if __has_builtin(__builtin_amdgcn_cvt_pkrtz)
  const v2hf p0 = __builtin_amdgcn_cvt_pkrtz(a0[0], a0[1]);
  const v2hf p1 = __builtin_amdgcn_cvt_pkrtz(a0[2], a0[3]);
  const v2hf p2 = __builtin_amdgcn_cvt_pkrtz(a1[0], a1[1]);
  const v2hf p3 = __builtin_amdgcn_cvt_pkrtz(a1[2], a1[3]);
  v8h h;
  h[0] = (_Float16)p0[0]; h[1] = (_Float16)p0[1];
  h[2] = (_Float16)p1[0]; h[3] = (_Float16)p1[1];
  h[4] = (_Float16)p2[0]; h[5] = (_Float16)p2[1];
  h[6] = (_Float16)p3[0]; h[7] = (_Float16)p3[1];
  return h;
#else
  v8h h;
#pragma unroll
  for (int i = 0; i < 4; ++i) { h[i] = (_Float16)a0[i]; h[4 + i] = (_Float16)a1[i]; }
  return h;
#endif
}

// ----------------------------- WMMA fragment IO ----------------------------
// Fragment layouts used by this code (A/B in contiguous K order, not the ISA table order; C/D per ISA):
//   A (16x32 f16): lane l: m=l&15, sel=l>>4; elem j: K = 16*sel + j
//   B (32x16 f16): lane l: n=l&15, sel=l>>4; elem j: K = 16*sel + j
//   C/D (16x16 f32): vgpr r: m = r + 8*sel, n = lane&15

static __device__ __forceinline__ v16h load_frag_lm(const _Float16* base) {
  const int l = threadIdx.x & 31;
  const v8h* p = (const v8h*)(base + l * 16);
  return cat8(p[0], p[1]);
}
static __device__ __forceinline__ v16h load_a_frag_g(const _Float16* p, int ld) {
  const int lane = threadIdx.x & 31;
  const int m = lane & 15, sel = lane >> 4;
  const v8h* q0 = (const v8h*)(p + (size_t)m * ld + 16 * sel);
  const v8h* q1 = (const v8h*)(p + (size_t)m * ld + 8 + 16 * sel);
  return cat8(*q0, *q1);
}
static __device__ __forceinline__ v16h load_b_frag_rowT_g(const _Float16* p, int ld) {
  const int lane = threadIdx.x & 31;
  const int n = lane & 15, sel = lane >> 4;
  const v8h* q = (const v8h*)(p + (size_t)n * ld + 16 * sel);
  return cat8(q[0], q[1]);
}
static __device__ __forceinline__ void store_c_frag(float* p, int ld, v8f c) {
  const int lane = threadIdx.x & 31;
  const int n = lane & 15, sel = lane >> 4;
#pragma unroll
  for (int r = 0; r < 8; ++r) p[(size_t)(r + 8 * sel) * ld + n] = c[r];
}
static __device__ __forceinline__ void store_c_frag_f16(_Float16* p, int ld, v8f c) {
  const int lane = threadIdx.x & 31;
  const int n = lane & 15, sel = lane >> 4;
#pragma unroll
  for (int r = 0; r < 8; ++r) p[(size_t)(r + 8 * sel) * ld + n] = (_Float16)c[r];
}

// ---- swizzled f32 state S in LDS: slot(rb,cs,l,j) = S[rb*32+16*(l>>4)+j][cs*16+(l&15)]
static __device__ __forceinline__ v16h load_b_frag_S(const float* Sb, int rb, int cs) {
  const int l = threadIdx.x & 31;
  const f4v* p = (const f4v*)(Sb + ((size_t)((rb * 16 + cs) * 32 + l)) * 16);
  return cat8(pk8(p[0], p[1]), pk8(p[2], p[3]));
}
static __device__ __forceinline__ float* cfrag_S_ptr(float* Sb, int mt, int nt) {
  const int lane = threadIdx.x & 31;
  const int nl = lane & 15, sel = lane >> 4;
  return Sb + ((size_t)(((mt >> 1) * 16 + nt) * 32 + nl + 16 * (mt & 1))) * 16 + 8 * sel;
}
static __device__ __forceinline__ v8f load_c_frag_S(float* Sb, int mt, int nt) {
  const f4v* p = (const f4v*)cfrag_S_ptr(Sb, mt, nt);
  const f4v f0 = p[0], f1 = p[1];
  v8f c;
#pragma unroll
  for (int i = 0; i < 4; ++i) { c[i] = f0[i]; c[4 + i] = f1[i]; }
  return c;
}
static __device__ __forceinline__ void store_c_frag_S(float* Sb, int mt, int nt, v8f c) {
  f4v* p = (f4v*)cfrag_S_ptr(Sb, mt, nt);
  f4v f0, f1;
#pragma unroll
  for (int i = 0; i < 4; ++i) { f0[i] = c[i]; f1[i] = c[4 + i]; }
  p[0] = f0; p[1] = f1;
}

// ------------------------- generic WMMA GEMM (f32 io) ----------------------
// C[M,N] = act(A[M,K]*B[K,N] + bias).  Block tile 128x128, K-step 32, 8 waves.
// N compile-time (B strides fold to immediates); K runtime (touches only the
// one-time pointer setup -- staging pointers advance by constant strides).
// REQUIRES: M % 128 == 0, N % 128 == 0, K % 8 == 0 (tail guarded per 8-group).
template <int N>
__global__ __launch_bounds__(256) void k_wmma_gemm(const float* __restrict__ A,
                                                   const float* __restrict__ B,
                                                   float* __restrict__ C,
                                                   int K,
                                                   const float* __restrict__ bias,
                                                   int act) {
  __shared__ __align__(16) _Float16 Asw[8 * 32 * 16];   // 8 m-subtiles, lane-major
  __shared__ __align__(16) _Float16 Bsw[8 * 32 * 16];   // 8 n-subtiles, lane-major
  const int tid = threadIdx.x;
  const int m0 = blockIdx.y * 128, n0 = blockIdx.x * 128;
  const int w = tid >> 5;
  const int wm = (w >> 1) * 32;
  const int wn = (w & 1) * 64;
  v8f acc[2][4] = {};
  const int kfull = K >> 5;
  // loop-invariant staging pointers (advance by constant strides)
  const int mA0 = tid >> 2,          kg0 = tid & 3;
  const int mA1 = (tid + 256) >> 2,  kg1 = (tid + 256) & 3;
  const float* ap0 = A + (size_t)(m0 + mA0) * K + kg0 * 8;
  const float* ap1 = A + (size_t)(m0 + mA1) * K + kg1 * 8;
  _Float16* ad0 = &Asw[(size_t)(((mA0 >> 4) * 32) + (mA0 & 15) + 16 * (kg0 >> 1)) * 16 + (kg0 & 1) * 8];
  _Float16* ad1 = &Asw[(size_t)(((mA1 >> 4) * 32) + (mA1 & 15) + 16 * (kg1 >> 1)) * 16 + (kg1 & 1) * 8];
  const int cB = tid & 127, selB = tid >> 7;
  const float* bp = B + (size_t)(16 * selB) * N + n0 + cB;
  _Float16* bd = &Bsw[(size_t)(((cB >> 4) * 32) + (cB & 15) + 16 * selB) * 16];
  // ------------------ branch-free main K loop (full 32-steps) ------------------
  for (int ks = 0; ks < kfull; ++ks) {
    *(v8h*)ad0 = pk8(*(const f4v*)ap0, *(const f4v*)(ap0 + 4));
    *(v8h*)ad1 = pk8(*(const f4v*)ap1, *(const f4v*)(ap1 + 4));
    {
      f4v x0, x1, x2, x3;
#pragma unroll
      for (int j = 0; j < 4; ++j) {
        x0[j] = bp[(size_t)j * N];
        x1[j] = bp[(size_t)(j + 4) * N];
        x2[j] = bp[(size_t)(j + 8) * N];
        x3[j] = bp[(size_t)(j + 12) * N];
      }
      *(v8h*)bd = pk8(x0, x1);
      *(v8h*)(bd + 8) = pk8(x2, x3);
    }
    if (ks + 1 < kfull) {   // speculative prefetch of next K tiles
      __builtin_prefetch(ap0 + 32, 0, 1);
      __builtin_prefetch(bp + (size_t)32 * N, 0, 1);
    }
    __syncthreads();
    v16h af[2], bf[4];
#pragma unroll
    for (int i = 0; i < 2; ++i) af[i] = load_frag_lm(&Asw[(size_t)((wm >> 4) + i) * 512]);
#pragma unroll
    for (int j = 0; j < 4; ++j) bf[j] = load_frag_lm(&Bsw[(size_t)((wn >> 4) + j) * 512]);
#pragma unroll
    for (int i = 0; i < 2; ++i)
#pragma unroll
      for (int j = 0; j < 4; ++j) acc[i][j] = wmma16(af[i], bf[j], acc[i][j]);
    __syncthreads();
    ap0 += 32; ap1 += 32; bp += (size_t)32 * N;
  }
  // ------------------ hoisted K tail (at most one 32-round) -------------------
  if (kfull * 32 < K) {
    const int k0 = kfull * 32;
    v8h hA0 = {}, hA1 = {};
    if (k0 + kg0 * 8 < K) hA0 = pk8(*(const f4v*)ap0, *(const f4v*)(ap0 + 4));
    if (k0 + kg1 * 8 < K) hA1 = pk8(*(const f4v*)ap1, *(const f4v*)(ap1 + 4));
    *(v8h*)ad0 = hA0;
    *(v8h*)ad1 = hA1;
    {
      v8h h0 = {}, h1 = {};
      if (k0 + 16 * selB < K) {          // rows 16*selB .. +7 valid as a group
        f4v x0, x1;
#pragma unroll
        for (int j = 0; j < 4; ++j) {
          x0[j] = bp[(size_t)j * N];
          x1[j] = bp[(size_t)(j + 4) * N];
        }
        h0 = pk8(x0, x1);
      }
      if (k0 + 16 * selB + 8 < K) {      // rows 16*selB+8 .. +15 valid as a group
        f4v x2, x3;
#pragma unroll
        for (int j = 0; j < 4; ++j) {
          x2[j] = bp[(size_t)(j + 8) * N];
          x3[j] = bp[(size_t)(j + 12) * N];
        }
        h1 = pk8(x2, x3);
      }
      *(v8h*)bd = h0;
      *(v8h*)(bd + 8) = h1;
    }
    __syncthreads();
    v16h af[2], bf[4];
#pragma unroll
    for (int i = 0; i < 2; ++i) af[i] = load_frag_lm(&Asw[(size_t)((wm >> 4) + i) * 512]);
#pragma unroll
    for (int j = 0; j < 4; ++j) bf[j] = load_frag_lm(&Bsw[(size_t)((wn >> 4) + j) * 512]);
#pragma unroll
    for (int i = 0; i < 2; ++i)
#pragma unroll
      for (int j = 0; j < 4; ++j) acc[i][j] = wmma16(af[i], bf[j], acc[i][j]);
  }
  // --------------------------------- epilogue --------------------------------
  const int lane = tid & 31;
  const int nl = lane & 15, sel = lane >> 4;
#pragma unroll
  for (int i = 0; i < 2; ++i)
#pragma unroll
    for (int j = 0; j < 4; ++j) {
      const int gn = n0 + wn + j * 16 + nl;
      const float bv = bias ? bias[gn] : 0.f;
#pragma unroll
      for (int r = 0; r < 8; ++r) {
        const int gm = m0 + wm + i * 16 + r + 8 * sel;
        float v = acc[i][j][r] + bv;
        if (act == 1) v = geluf(v);
        C[(size_t)gm * N + gn] = v;
      }
    }
}

// ----------------------- causal depthwise conv kernels ---------------------
__global__ void k_conv4_silu(const float* __restrict__ x, const float* __restrict__ wv,
                             float* __restrict__ y, int total) {
  int i = blockIdx.x * blockDim.x + threadIdx.x;
  if (i >= total) return;
  const int c = i & (D_ - 1);
  const int l = (i >> 10) & (L_ - 1);
  const int b = i >> 22;
  float acc = 0.f;
  if (l >= 3) {
    const float* p = x + ((size_t)(b * L_ + l - 3)) * D_ + c;
#pragma unroll
    for (int t = 0; t < 4; ++t) acc += wv[c * 4 + t] * p[(size_t)t * D_];
  } else {
#pragma unroll
    for (int t = 0; t < 4; ++t) {
      const int ls = l - 3 + t;
      if (ls >= 0) acc += wv[c * 4 + t] * x[((size_t)(b * L_ + ls)) * D_ + c];
    }
  }
  y[i] = siluf(acc);
}

__global__ void k_fir(const float* __restrict__ x, const float* __restrict__ wv,
                      float* __restrict__ y, int K, int total) {
  int i = blockIdx.x * blockDim.x + threadIdx.x;
  if (i >= total) return;
  const int c = i & (D_ - 1);
  const int l = (i >> 10) & (L_ - 1);
  const int b = i >> 22;
  float acc = 0.f;
  if (l >= K - 1) {
    const float* p = x + ((size_t)(b * L_ + l - (K - 1))) * D_ + c;
    for (int t = 0; t < K; ++t) acc += wv[c * K + t] * p[(size_t)t * D_];
  } else {
    for (int t = 0; t < K; ++t) {
      const int ls = l - (K - 1) + t;
      if (ls >= 0) acc += wv[c * K + t] * x[((size_t)(b * L_ + ls)) * D_ + c];
    }
  }
  y[i] = acc;
}

// ---------------- per-row 4-wide GEMVs (one thread per (b,l)) --------------
__global__ void k_beta(const float* __restrict__ hs, const float* __restrict__ Wb,
                       float* __restrict__ beta, int total) {
  int bl = blockIdx.x * blockDim.x + threadIdx.x;
  if (bl >= total) return;
  const f4v* hp = (const f4v*)(hs + (size_t)bl * D_);
  const f4v* wp = (const f4v*)Wb;
  f4v acc = {};
  for (int k4 = 0; k4 < D_ / 4; ++k4) {
    const f4v hv = hp[k4];
#pragma unroll
    for (int i = 0; i < 4; ++i) acc += wp[k4 * 4 + i] * hv[i];
  }
#pragma unroll
  for (int j = 0; j < 4; ++j) beta[(size_t)bl * 4 + j] = sigmf(acc[j]);
}

__global__ void k_resgate(const float* __restrict__ hs, const float* __restrict__ Wres,
                          const float* __restrict__ bres, const float* __restrict__ crl,
                          float* __restrict__ rg, int total) {
  int bl = blockIdx.x * blockDim.x + threadIdx.x;
  if (bl >= total) return;
  const f4v* hp = (const f4v*)(hs + (size_t)bl * D_);
  const f4v* wp = (const f4v*)Wres;
  f4v acc = {};
  for (int k4 = 0; k4 < D_ / 4; ++k4) {
    const f4v hv = hp[k4];
#pragma unroll
    for (int i = 0; i < 4; ++i) acc += wp[k4 * 4 + i] * hv[i];
  }
#pragma unroll
  for (int j = 0; j < 4; ++j)
    rg[(size_t)bl * 4 + j] = sigmf(crl[j]) * sigmf(acc[j] + bres[j]);
}

__global__ void k_logits(const float* __restrict__ g1, const float* __restrict__ Wg2,
                         const float* __restrict__ bg2, const float* __restrict__ temp,
                         float* __restrict__ out, int total) {
  int bl = blockIdx.x * blockDim.x + threadIdx.x;
  if (bl >= total) return;
  const f4v* gp = (const f4v*)(g1 + (size_t)bl * D_);
  const f4v* wp = (const f4v*)Wg2;
  f4v acc = {};
  for (int k4 = 0; k4 < D_ / 4; ++k4) {
    const f4v gv = gp[k4];
#pragma unroll
    for (int i = 0; i < 4; ++i) acc += wp[k4 * 4 + i] * gv[i];
  }
  const float inv_t = 1.f / log1pf(__expf(temp[0]));
#pragma unroll
  for (int j = 0; j < 4; ++j) out[(size_t)bl * 4 + j] = (acc[j] + bg2[j]) * inv_t;
}

__global__ void k_probs(const float* __restrict__ lg, float* __restrict__ p, int total) {
  int bl = blockIdx.x * blockDim.x + threadIdx.x;
  if (bl >= total) return;
  float v[4], m = -1e30f;
#pragma unroll
  for (int j = 0; j < 4; ++j) { v[j] = lg[bl * 4 + j]; m = fmaxf(m, v[j]); }
  float s = 0.f;
#pragma unroll
  for (int j = 0; j < 4; ++j) { v[j] = __expf(v[j] - m); s += v[j]; }
  float s2 = 0.f;
#pragma unroll
  for (int j = 0; j < 4; ++j) { v[j] = fmaxf(v[j] / s, 0.02f); s2 += v[j]; }
#pragma unroll
  for (int j = 0; j < 4; ++j) p[bl * 4 + j] = v[j] / s2;
}

__global__ void k_og2(const float* __restrict__ og1, const float* __restrict__ Wog2,
                      const float* __restrict__ bog2, float* __restrict__ og, int total) {
  int bl = blockIdx.x * blockDim.x + threadIdx.x;
  if (bl >= total) return;
  const f4v* gp = (const f4v*)(og1 + (size_t)bl * 512);
  const f4v* wp = (const f4v*)Wog2;
  f4v acc = {};
  for (int k4 = 0; k4 < 128; ++k4) acc += gp[k4] * wp[k4];
  og[bl] = sigmf(acc[0] + acc[1] + acc[2] + acc[3] + bog2[0]);
}

// ---------------------- delta-rule prep: l2norm + scale --------------------
__global__ __launch_bounds__(256) void k_prep_norm(const float* __restrict__ qc,
                                                   const float* __restrict__ kc,
                                                   const float* __restrict__ vc,
                                                   const float* __restrict__ beta,
                                                   _Float16* __restrict__ q16,
                                                   _Float16* __restrict__ k16,
                                                   _Float16* __restrict__ kT16,
                                                   _Float16* __restrict__ v16,
                                                   _Float16* __restrict__ kb16) {
  __shared__ float red[256];
  const int tid = threadIdx.x;
  const int l = blockIdx.x & (L_ - 1);
  const int bh = blockIdx.x >> 12;
  const int b = bh >> 2, h = bh & 3;
  const size_t src = ((size_t)(b * L_ + l)) * D_ + h * DV_ + tid;
  const float qv = qc[src], kv = kc[src], vv = vc[src];
  const float bet = beta[(size_t)(b * L_ + l) * H_ + h];
  red[tid] = qv * qv; __syncthreads();
  for (int s = 128; s > 0; s >>= 1) { if (tid < s) red[tid] += red[tid + s]; __syncthreads(); }
  const float rq = rsqrtf(red[0] + 1e-6f); __syncthreads();
  red[tid] = kv * kv; __syncthreads();
  for (int s = 128; s > 0; s >>= 1) { if (tid < s) red[tid] += red[tid + s]; __syncthreads(); }
  const float rk = rsqrtf(red[0] + 1e-6f);
  const float qn = qv * rq, kn = kv * rk;
  const size_t di = ((size_t)bh * L_ + l) * DV_ + tid;
  q16[di] = (_Float16)qn;
  k16[di] = (_Float16)kn;
  kT16[((size_t)bh * DV_ + tid) * L_ + l] = (_Float16)kn;
  v16[di]  = (_Float16)(vv * bet);
  kb16[di] = (_Float16)(kn * bet);
}

// ------------------- delta-rule prep: per-chunk UT transform ---------------
__global__ __launch_bounds__(128) void k_prep_chunk(const _Float16* __restrict__ k16,
                                                    const _Float16* __restrict__ v16,
                                                    const _Float16* __restrict__ kb16,
                                                    float* __restrict__ u,
                                                    _Float16* __restrict__ w16) {
  __shared__ __align__(16) _Float16 vt[16 * 32 * 16];   // B-frag-major
  __shared__ __align__(16) _Float16 kbt[16 * 32 * 16];  // B-frag-major
  __shared__ float    invf[32 * 33];
  __shared__ __align__(16) _Float16 invh[2 * 32 * 16];  // A-frag-major
  const int tid = threadIdx.x;
  const int w = tid >> 5;
  const int lane = tid & 31;
  const int nl = lane & 15, sel = lane >> 4;
  const int bh = blockIdx.x >> 7;
  const int l0 = (blockIdx.x & 127) * CHK_;
  const _Float16* kg  = k16  + (size_t)bh * L_ * DV_;
  const _Float16* vg  = v16  + (size_t)bh * L_ * DV_;
  const _Float16* kbg = kb16 + (size_t)bh * L_ * DV_;
  for (int s4 = 0; s4 < 4; ++s4) {
    const int slot = tid + s4 * 128;
    const int cs = slot >> 5, ls = slot & 31;
    const int n = cs * 16 + (ls & 15);
    const int kb2 = 16 * (ls >> 4);
    v8h v0, v1, w0, w1;
#pragma unroll
    for (int j = 0; j < 16; ++j) {
      const size_t gi = (size_t)(l0 + kb2 + j) * DV_ + n;
      const _Float16 vv = vg[gi], kv = kbg[gi];
      if (j < 8) { v0[j] = vv; w0[j] = kv; } else { v1[j - 8] = vv; w1[j - 8] = kv; }
    }
    _Float16* dv = &vt[(size_t)slot * 16];
    _Float16* dk = &kbt[(size_t)slot * 16];
    *(v8h*)dv = v0; *(v8h*)(dv + 8) = v1;
    *(v8h*)dk = w0; *(v8h*)(dk + 8) = w1;
  }
  __syncthreads();
  {
    const int mt = w >> 1, nt = w & 1;
    v8f acc = {};
#pragma unroll
    for (int ks = 0; ks < 8; ++ks) {
      v16h a = load_a_frag_g(kbg + (size_t)(l0 + mt * 16) * DV_ + ks * 32, DV_);
      v16h b = load_b_frag_rowT_g(kg + (size_t)(l0 + nt * 16) * DV_ + ks * 32, DV_);
      acc = wmma16(a, b, acc);
    }
#pragma unroll
    for (int r = 0; r < 8; ++r) {
      const int mg = mt * 16 + r + 8 * sel;
      const int ng = nt * 16 + nl;
      invf[mg * 33 + ng] = (mg > ng) ? -acc[r] : 0.f;
    }
  }
  __syncthreads();
  for (int i = 1; i < 32; ++i) {
    if (tid < 32) {
      float s = 0.f;
      for (int j = 0; j < i; ++j) s += invf[i * 33 + j] * invf[j * 33 + tid];
      invf[i * 33 + tid] += s;
    }
    __syncthreads();
  }
  for (int i = tid; i < 1024; i += 128) {
    const int m = i >> 5, c = i & 31;
    const float val = invf[m * 33 + c] + (m == c ? 1.f : 0.f);
    invh[(size_t)((m >> 4) * 32 + (m & 15) + 16 * (c >> 4)) * 16 + (c & 15)]
        = (_Float16)val;
  }
  __syncthreads();
  {
    const int mt = w >> 1;
    v16h a = load_frag_lm(&invh[(size_t)mt * 512]);
#pragma unroll
    for (int it = 0; it < 8; ++it) {
      const int nt = (w & 1) * 8 + it;
      v8f cu = {};
      cu = wmma16(a, load_frag_lm(&vt[(size_t)nt * 512]), cu);
      store_c_frag(u + ((size_t)bh * L_ + l0 + mt * 16) * DV_ + nt * 16, DV_, cu);
      v8f cw = {};
      cw = wmma16(a, load_frag_lm(&kbt[(size_t)nt * 512]), cw);
      store_c_frag_f16(w16 + ((size_t)bh * L_ + l0 + mt * 16) * DV_ + nt * 16, DV_, cw);
    }
  }
}

// -------------------- delta-rule sequential chunk scan ---------------------
#define SCAN_LDS_BYTES (256 * 256 * 4 + 16 * 32 * 16 * 2 + 2 * 32 * 16 * 2)
__global__ __launch_bounds__(512) void k_scan(const _Float16* __restrict__ q16,
                                              const _Float16* __restrict__ k16,
                                              const _Float16* __restrict__ kT16,
                                              const _Float16* __restrict__ w16,
                                              const float* __restrict__ u,
                                              float* __restrict__ dout) {
  extern __shared__ char smem[];
  float*    S     = (float*)smem;
  _Float16* ui2h  = (_Float16*)(smem + 256 * 256 * 4);
  _Float16* attnh = (_Float16*)(smem + 256 * 256 * 4 + 16 * 32 * 16 * 2);
  const int tid = threadIdx.x;
  const int w = tid >> 5;
  const int lane = tid & 31;
  const int nl = lane & 15, sel = lane >> 4;
  const int bh = blockIdx.x;
  const int b = bh >> 2, h = bh & 3;
  const _Float16* qg  = q16  + (size_t)bh * L_ * DV_;
  const _Float16* kg  = k16  + (size_t)bh * L_ * DV_;
  const _Float16* ktg = kT16 + (size_t)bh * DV_ * L_;
  const _Float16* wg  = w16  + (size_t)bh * L_ * DV_;
  const float*    ug  = u    + (size_t)bh * L_ * DV_;
  for (int i = tid; i < 256 * 256; i += 512) S[i] = 0.f;
  __syncthreads();
  for (int ic = 0; ic < NC_; ++ic) {
    const int l0 = ic * CHK_;
    if (ic + 1 < NC_) {
      const size_t nb = (size_t)(l0 + CHK_) * DV_ + tid * 16;
      __builtin_prefetch(qg + nb, 0, 1);
      __builtin_prefetch(wg + nb, 0, 1);
      __builtin_prefetch(ug + nb, 0, 1);
    }
    // ---- ui2 = u - w @ S ----
#pragma unroll
    for (int s2 = 0; s2 < 2; ++s2) {
      const int t = w * 2 + s2;
      const int mt = t >> 4, nt = t & 15;
      v8f acc = {};
#pragma unroll
      for (int ks = 0; ks < 8; ++ks) {
        v16h a = load_a_frag_g(wg + (size_t)(l0 + mt * 16) * DV_ + ks * 32, DV_);
        acc = wmma16(a, load_b_frag_S(S, ks, nt), acc);
      }
      const float* up = ug + (size_t)(l0 + mt * 16) * DV_ + nt * 16;
      f4v x0, x1;
#pragma unroll
      for (int r = 0; r < 4; ++r) {
        x0[r] = up[(size_t)(r + 8 * sel) * DV_ + nl] - acc[r];
        x1[r] = up[(size_t)(r + 4 + 8 * sel) * DV_ + nl] - acc[r + 4];
      }
      *(v8h*)(ui2h + (size_t)(nt * 32 + nl + 16 * mt) * 16 + 8 * sel) = pk8(x0, x1);
    }
    __syncthreads();
    // ---- attn = tril_incl(q @ k^T) ----
    if (w < 4) {
      const int mt = w >> 1, nt = w & 1;
      v8f acc = {};
#pragma unroll
      for (int ks = 0; ks < 8; ++ks) {
        v16h a = load_a_frag_g(qg + (size_t)(l0 + mt * 16) * DV_ + ks * 32, DV_);
        v16h bfr = load_b_frag_rowT_g(kg + (size_t)(l0 + nt * 16) * DV_ + ks * 32, DV_);
        acc = wmma16(a, bfr, acc);
      }
#pragma unroll
      for (int r = 0; r < 8; ++r) {
        const int mg = mt * 16 + r + 8 * sel;
        const int ng = nt * 16 + nl;
        attnh[(size_t)(mt * 32 + (mg & 15) + 16 * (ng >> 4)) * 16 + (ng & 15)]
            = (_Float16)((mg >= ng) ? acc[r] : 0.f);
      }
    }
    __syncthreads();
    // ---- o = q @ S + attn @ ui2 ----
#pragma unroll
    for (int s2 = 0; s2 < 2; ++s2) {
      const int t = w * 2 + s2;
      const int mt = t >> 4, nt = t & 15;
      v8f acc = {};
#pragma unroll
      for (int ks = 0; ks < 8; ++ks) {
        v16h a = load_a_frag_g(qg + (size_t)(l0 + mt * 16) * DV_ + ks * 32, DV_);
        acc = wmma16(a, load_b_frag_S(S, ks, nt), acc);
      }
      v16h aa = load_frag_lm(attnh + (size_t)mt * 512);
      v16h bb = load_frag_lm(ui2h + (size_t)nt * 512);
      acc = wmma16(aa, bb, acc);
      float* op = dout + ((size_t)(b * L_ + l0 + mt * 16) * H_ + h) * DV_ + nt * 16;
      store_c_frag(op, H_ * DV_, acc);
    }
    __syncthreads();
    // ---- S += k^T @ ui2 ----
    {
      v16h a = load_a_frag_g(ktg + (size_t)(w * 16) * L_ + l0, L_);
#pragma unroll
      for (int i5 = 0; i5 < 16; ++i5) {
        v8f c = load_c_frag_S(S, w, i5);
        c = wmma16(a, load_frag_lm(ui2h + (size_t)i5 * 512), c);
        store_c_frag_S(S, w, i5, c);
      }
    }
    __syncthreads();
  }
}

// ------------------------------ stats & gating -----------------------------
__global__ __launch_bounds__(256) void k_stats4(const float* __restrict__ src,
                                                float* __restrict__ out,
                                                int ldout, int off) {
  __shared__ float r1[256], r2[256], r3[256];
  const int tid = threadIdx.x;
  const int bl = blockIdx.x;
  float a0 = 0.f, a1 = 0.f, a2 = 0.f, a3 = 0.f;
  for (int h = 0; h < H_; ++h) {
    const float x = src[(size_t)bl * D_ + h * DV_ + tid];
    r1[tid] = x; r2[tid] = x * x; r3[tid] = fabsf(x);
    __syncthreads();
    for (int s = 128; s > 0; s >>= 1) {
      if (tid < s) { r1[tid] += r1[tid + s]; r2[tid] += r2[tid + s]; r3[tid] += r3[tid + s]; }
      __syncthreads();
    }
    const float mean = r1[0] * (1.f / 256.f);
    const float sq   = r2[0];
    a0 += mean * 0.25f;
    a1 += (sq * (1.f / 256.f) - mean * mean) * 0.25f;
    a2 += r3[0] * (1.f / 256.f) * 0.25f;
    a3 += sqrtf(sq) * 0.25f;
    __syncthreads();
  }
  if (tid == 0) {
    float* o = out + (size_t)bl * ldout + off;
    o[0] = a0; o[1] = a1; o[2] = a2; o[3] = a3;
  }
}

__global__ void k_gate_cat(const float* __restrict__ hs, const float* __restrict__ stats,
                           float* __restrict__ out, int total) {
  int i = blockIdx.x * blockDim.x + threadIdx.x;
  if (i >= total) return;
  const int bl = i / 1040, c = i % 1040;
  out[i] = (c < 1024) ? hs[(size_t)bl * D_ + c] : stats[(size_t)bl * 16 + (c - 1024)];
}

__global__ void k_og_cat(const float* __restrict__ hs, const float* __restrict__ ostats,
                         const float* __restrict__ stats, float* __restrict__ out, int total) {
  int i = blockIdx.x * blockDim.x + threadIdx.x;
  if (i >= total) return;
  const int bl = i / 1032, c = i % 1032;
  float v;
  if (c < 1024)      v = hs[(size_t)bl * D_ + c];
  else if (c < 1028) v = ostats[(size_t)bl * 4 + (c - 1024)];
  else               v = stats[(size_t)bl * 16 + 12 + (c - 1028)];
  out[i] = v;
}

__global__ void k_combine(const float* __restrict__ ls, const float* __restrict__ ll,
                          const float* __restrict__ dd, const float* __restrict__ vd,
                          const float* __restrict__ p, const float* __restrict__ rg,
                          float* __restrict__ o, int total) {
  int i = blockIdx.x * blockDim.x + threadIdx.x;
  if (i >= total) return;
  const int c = i & (D_ - 1);
  const int bl = i >> 10;
  const int h = c >> 8;
  const float* pp = p + (size_t)bl * 4;
  o[i] = pp[0] * ls[i] + pp[1] * ll[i] + pp[2] * dd[i] + pp[3] * vd[i]
       + rg[(size_t)bl * H_ + h] * ls[i];
}

__global__ __launch_bounds__(256) void k_scale_rmsnorm(const float* __restrict__ o,
                                                       const float* __restrict__ og,
                                                       const float* __restrict__ nw,
                                                       float* __restrict__ on) {
  __shared__ float red[256];
  const int tid = threadIdx.x;
  const int blh = blockIdx.x;
  const int bl = blh >> 2, h = blh & 3;
  const size_t idx = (size_t)bl * D_ + h * DV_ + tid;
  const float x = o[idx] * og[bl];
  red[tid] = x * x; __syncthreads();
  for (int s = 128; s > 0; s >>= 1) { if (tid < s) red[tid] += red[tid + s]; __syncthreads(); }
  on[idx] = x * rsqrtf(red[0] * (1.f / 256.f) + 1e-5f) * nw[tid];
}

// ------------------------------- host driver -------------------------------
extern "C" void kernel_launch(void* const* d_in, const int* in_sizes, int n_in,
                              void* d_out, int out_size, void* d_ws, size_t ws_size,
                              hipStream_t stream) {
  (void)in_sizes; (void)n_in; (void)out_size; (void)ws_size;
  const float* hs       = (const float*)d_in[0];
  const float* Wq       = (const float*)d_in[1];
  const float* Wk       = (const float*)d_in[2];
  const float* Wv       = (const float*)d_in[3];
  const float* Wb       = (const float*)d_in[4];
  const float* wq_conv  = (const float*)d_in[5];
  const float* wk_conv  = (const float*)d_in[6];
  const float* wv_conv  = (const float*)d_in[7];
  const float* fir_long = (const float*)d_in[8];
  const float* fir_short= (const float*)d_in[9];
  const float* Wg1      = (const float*)d_in[10];
  const float* bg1      = (const float*)d_in[11];
  const float* Wg2      = (const float*)d_in[12];
  const float* bg2      = (const float*)d_in[13];
  const float* ltemp    = (const float*)d_in[14];
  const float* crl      = (const float*)d_in[15];
  const float* Wres     = (const float*)d_in[16];
  const float* bres     = (const float*)d_in[17];
  const float* Wog1     = (const float*)d_in[18];
  const float* bog1     = (const float*)d_in[19];
  const float* Wog2     = (const float*)d_in[20];
  const float* bog2     = (const float*)d_in[21];
  const float* norm_w   = (const float*)d_in[22];
  const float* Wo       = (const float*)d_in[23];
  float* out = (float*)d_out;

  char* W = (char*)d_ws; size_t used = 0;
  auto ALLOC = [&](size_t bytes) -> void* {
    void* p = (void*)(W + used);
    used = (used + bytes + 255) & ~(size_t)255;
    return p;
  };
  float* q_lin  = (float*)ALLOC(BLD_ * 4);
  float* k_lin  = (float*)ALLOC(BLD_ * 4);
  float* v_lin  = (float*)ALLOC(BLD_ * 4);
  float* q_c    = (float*)ALLOC(BLD_ * 4);
  float* k_c    = (float*)ALLOC(BLD_ * 4);
  float* v_c    = (float*)ALLOC(BLD_ * 4);
  float* beta   = (float*)ALLOC((size_t)BL_ * H_ * 4);
  float* u      = (float*)ALLOC(BLD_ * 4);
  float* dout   = (float*)ALLOC(BLD_ * 4);
  float* ls     = (float*)ALLOC(BLD_ * 4);
  float* llb    = (float*)ALLOC(BLD_ * 4);
  float* stats  = (float*)ALLOC((size_t)BL_ * 16 * 4);
  float* gatein = (float*)ALLOC((size_t)BL_ * 1040 * 4);
  float* g1     = (float*)ALLOC(BLD_ * 4);
  float* logits = (float*)ALLOC((size_t)BL_ * 4 * 4);
  float* probs  = (float*)ALLOC((size_t)BL_ * 4 * 4);
  float* rg     = (float*)ALLOC((size_t)BL_ * H_ * 4);
  float* ostats = (float*)ALLOC((size_t)BL_ * 4 * 4);
  float* ogin   = (float*)ALLOC((size_t)BL_ * 1032 * 4);
  float* og1    = (float*)ALLOC((size_t)BL_ * 512 * 4);
  float* og     = (float*)ALLOC((size_t)BL_ * 4);
  float* obuf   = (float*)ALLOC(BLD_ * 4);
  float* onorm  = (float*)ALLOC(BLD_ * 4);
  _Float16* q16  = (_Float16*)ALLOC(BLD_ * 2);
  _Float16* k16  = (_Float16*)ALLOC(BLD_ * 2);
  _Float16* kT16 = (_Float16*)ALLOC(BLD_ * 2);
  _Float16* v16  = (_Float16*)ALLOC(BLD_ * 2);
  _Float16* kb16 = (_Float16*)ALLOC(BLD_ * 2);
  _Float16* w16  = (_Float16*)ALLOC(BLD_ * 2);

  const dim3 gNd(8, 64);    // N=1024 tiles
  const dim3 gNh(4, 64);    // N=512 tiles

  k_wmma_gemm<1024><<<gNd, 256, 0, stream>>>(hs, Wq, q_lin, D_, nullptr, 0);
  k_wmma_gemm<1024><<<gNd, 256, 0, stream>>>(hs, Wk, k_lin, D_, nullptr, 0);
  k_wmma_gemm<1024><<<gNd, 256, 0, stream>>>(hs, Wv, v_lin, D_, nullptr, 0);
  const int ebl = (int)(BLD_ / 256);
  k_conv4_silu<<<ebl, 256, 0, stream>>>(q_lin, wq_conv, q_c, (int)BLD_);
  k_conv4_silu<<<ebl, 256, 0, stream>>>(k_lin, wk_conv, k_c, (int)BLD_);
  k_conv4_silu<<<ebl, 256, 0, stream>>>(v_lin, wv_conv, v_c, (int)BLD_);
  k_beta<<<(BL_ + 255) / 256, 256, 0, stream>>>(hs, Wb, beta, BL_);
  k_prep_norm<<<B_ * H_ * L_, 256, 0, stream>>>(q_c, k_c, v_c, beta, q16, k16, kT16, v16, kb16);
  k_prep_chunk<<<B_ * H_ * NC_, 128, 0, stream>>>(k16, v16, kb16, u, w16);
  (void)hipFuncSetAttribute((const void*)k_scan,
                            hipFuncAttributeMaxDynamicSharedMemorySize, SCAN_LDS_BYTES);
  k_scan<<<B_ * H_, 512, SCAN_LDS_BYTES, stream>>>(q16, k16, kT16, w16, u, dout);
  k_fir<<<ebl, 256, 0, stream>>>(v_c, fir_long, llb, 64, (int)BLD_);
  k_fir<<<ebl, 256, 0, stream>>>(v_c, fir_short, ls, 5, (int)BLD_);
  k_stats4<<<BL_, 256, 0, stream>>>(ls,   stats, 16, 0);
  k_stats4<<<BL_, 256, 0, stream>>>(llb,  stats, 16, 4);
  k_stats4<<<BL_, 256, 0, stream>>>(dout, stats, 16, 8);
  k_stats4<<<BL_, 256, 0, stream>>>(v_c,  stats, 16, 12);
  k_gate_cat<<<(BL_ * 1040 + 255) / 256, 256, 0, stream>>>(hs, stats, gatein, BL_ * 1040);
  k_wmma_gemm<1024><<<gNd, 256, 0, stream>>>(gatein, Wg1, g1, 1040, bg1, 1);
  k_logits<<<(BL_ + 255) / 256, 256, 0, stream>>>(g1, Wg2, bg2, ltemp, logits, BL_);
  k_probs<<<(BL_ + 255) / 256, 256, 0, stream>>>(logits, probs, BL_);
  k_resgate<<<(BL_ + 255) / 256, 256, 0, stream>>>(hs, Wres, bres, crl, rg, BL_);
  k_combine<<<ebl, 256, 0, stream>>>(ls, llb, dout, v_c, probs, rg, obuf, (int)BLD_);
  k_stats4<<<BL_, 256, 0, stream>>>(obuf, ostats, 4, 0);
  k_og_cat<<<(BL_ * 1032 + 255) / 256, 256, 0, stream>>>(hs, ostats, stats, ogin, BL_ * 1032);
  k_wmma_gemm<512><<<gNh, 256, 0, stream>>>(ogin, Wog1, og1, 1032, bog1, 1);
  k_og2<<<(BL_ + 255) / 256, 256, 0, stream>>>(og1, Wog2, bog2, og, BL_);
  k_scale_rmsnorm<<<BL_ * H_, 256, 0, stream>>>(obuf, og, norm_w, onorm);
  k_wmma_gemm<1024><<<gNd, 256, 0, stream>>>(onorm, Wo, out, D_, nullptr, 0);
}
